// SimpleHead_86749749445315
// MI455X (gfx1250) — matrix operand order repaired and checked by reading, not compiled, not run on hardware
//
#include <hip/hip_runtime.h>
#include <hip/hip_bf16.h>

typedef __attribute__((ext_vector_type(16))) __bf16 v16bf;
typedef __attribute__((ext_vector_type(8)))  __bf16 bf16x8;
typedef __attribute__((ext_vector_type(4)))  __bf16 bf16x4;
typedef __attribute__((ext_vector_type(8)))  float  v8f;
typedef __attribute__((ext_vector_type(4)))  float  f32x4;
typedef __attribute__((ext_vector_type(4)))  unsigned int u32x4;
typedef __attribute__((ext_vector_type(8)))  int    i32x8;
typedef __attribute__((ext_vector_type(4)))  int    i32x4;

// ---------------------------------------------------------------------------
// Convert fp32 matrix (rows x cin) to bf16 (rows x cout), zero padding cols.
// ---------------------------------------------------------------------------
__global__ __launch_bounds__(256)
void cvt_pad_bf16_k(const float* __restrict__ src, __bf16* __restrict__ dst,
                    long rows, int cin, int cout)
{
    long idx = (long)blockIdx.x * 256 + threadIdx.x;
    long total = rows * (long)cout;
    if (idx >= total) return;
    long r = idx / cout;
    int  c = (int)(idx - r * cout);
    float v = (c < cin) ? src[r * (long)cin + c] : 0.0f;
    dst[idx] = (__bf16)v;
}

__global__ void pad_bias_k(const float* __restrict__ b, float* __restrict__ out,
                           int n, int ntot)
{
    int i = blockIdx.x * blockDim.x + threadIdx.x;
    if (i < ntot) out[i] = (i < n) ? b[i] : 0.0f;
}

// ---------------------------------------------------------------------------
// WMMA GEMM: C[M x ncols] = A[M x 512](bf16) @ W[512 x ncols](bf16) + bias
// 256 threads = 8 waves. WG tile: 128 rows x 64 cols. Wave tile: 16 x 64.
// The full 512x64 B panel (64 KB) is DMA'd into LDS once by the Tensor Data
// Mover; the k-loop then runs with no barriers and no staging.
// grid = (M/128, ncols/64)
// ---------------------------------------------------------------------------
__global__ __launch_bounds__(256)
void gemm_bf16_wmma_k(const __bf16* __restrict__ A, const __bf16* __restrict__ W,
                      const float* __restrict__ bias, float* __restrict__ Cout,
                      int M, int ncols)
{
    __shared__ __align__(128) __bf16 ldsB[512 * 64];  // 64 KB: B rows 0..511, cols n0..n0+63

    const int tid  = threadIdx.x;
    const int wave = tid >> 5;
    const int lane = tid & 31;
    const int m0 = blockIdx.x * 128 + wave * 16;
    const int n0 = blockIdx.y * 64;

    // ---- TDM: load B panel W[0..511][n0..n0+63] -> LDS (wave 0 only) ------
    if (wave == 0) {
        unsigned lds_off = (unsigned)(unsigned long long)(const void*)ldsB;
        unsigned long long ga = (unsigned long long)(const void*)(W + (size_t)n0);
        const unsigned td0 = (unsigned)ncols;      // tensor dim0 (contiguous)
        const unsigned td1 = 512u;                 // tensor dim1 (rows)
        const unsigned tl0 = 64u;                  // tile dim0 (cols)
        const unsigned tl1 = 512u;                 // tile dim1 (rows)
        const unsigned long long s0 = (unsigned long long)ncols;        // dim0 stride
        const unsigned long long s1 = (unsigned long long)ncols * 512u; // dim1 stride (unused for 2D tile)

        u32x4 g0;
        g0[0] = 1u;                                                 // count=1, is_restore=0
        g0[1] = lds_off;                                            // lds_addr
        g0[2] = (unsigned)(ga & 0xffffffffull);                     // global_addr[31:0]
        g0[3] = (unsigned)((ga >> 32) & 0x1ffffffull) | (2u << 30); // global_addr[56:32] | type=2

        i32x8 g1;
        g1[0] = (int)(1u << 16);                                    // data_size=1 (2 bytes), mask=0
        g1[1] = (int)((td0 & 0xffffu) << 16);                       // tensor_dim0[15:0]
        g1[2] = (int)((td0 >> 16) | ((td1 & 0xffffu) << 16));       // tensor_dim0[31:16] | tensor_dim1[15:0]
        g1[3] = (int)((td1 >> 16) | (tl0 << 16));                   // tensor_dim1[31:16] | tile_dim0
        g1[4] = (int)(tl1);                                         // tile_dim1 | tile_dim2=0
        g1[5] = (int)(s0 & 0xffffffffull);                          // tensor_dim0_stride[31:0]
        g1[6] = (int)((s0 >> 32) | ((s1 & 0xffffull) << 16));       // stride0[47:32] | stride1[15:0]
        g1[7] = (int)((s1 >> 16) & 0xffffffffull);                  // stride1[47:16]

        i32x4 gz4 = {0, 0, 0, 0};
        i32x8 gz8 = {0, 0, 0, 0, 0, 0, 0, 0};
        __builtin_amdgcn_tensor_load_to_lds(g0, g1, gz4, gz4, gz8, 0);
        __builtin_amdgcn_s_wait_tensorcnt(0);
    }
    __syncthreads();

    v8f acc0 = {}, acc1 = {}, acc2 = {}, acc3 = {};

    // A-fragment addressing (16-bit A 16x32 layout):
    //   lane 0-15 : M = lane,     K pairs start at 0  (VGPR0..3) / 16 (VGPR4..7)
    //   lane 16-31: M = lane-16,  K pairs start at 8  / 24
    const int mrow = m0 + (lane & 15);
    const int kb   = (lane >> 4) * 8;
    const __bf16* __restrict__ arow = A + (size_t)mrow * 512;

    #pragma unroll 4
    for (int k0 = 0; k0 < 512; k0 += 32) {
        if (k0 + 64 < 512)
            __builtin_prefetch(arow + k0 + 64, 0, 3);   // global_prefetch_b8

        // --- A fragment: two 16B loads, pack into v16bf ---------------------
        bf16x8 alo = *(const bf16x8*)(arow + k0 + kb);
        bf16x8 ahi = *(const bf16x8*)(arow + k0 + kb + 16);
        v16bf a;
        #pragma unroll
        for (int i = 0; i < 8; ++i) { a[i] = alo[i]; a[8 + i] = ahi[i]; }

        // --- B fragments from LDS: lane l holds column N=l&15, K in same order as A ----
        v16bf b0, b1, b2, b3;
        #pragma unroll
        for (int i = 0; i < 16; ++i) {
            const __bf16* brow = &ldsB[(k0 + kb + i + (i >= 8 ? 8 : 0)) * 64 + (lane & 15)];
            b0[i] = brow[0];
            b1[i] = brow[16];
            b2[i] = brow[32];
            b3[i] = brow[48];
        }

        acc0 = __builtin_amdgcn_wmma_f32_16x16x32_bf16(false, a, false, b0, (short)0, acc0, false, false);
        acc1 = __builtin_amdgcn_wmma_f32_16x16x32_bf16(false, a, false, b1, (short)0, acc1, false, false);
        acc2 = __builtin_amdgcn_wmma_f32_16x16x32_bf16(false, a, false, b2, (short)0, acc2, false, false);
        acc3 = __builtin_amdgcn_wmma_f32_16x16x32_bf16(false, a, false, b3, (short)0, acc3, false, false);
    }

    // --- epilogue: C/D layout: VGPR r -> M = r (+8 for lanes 16-31), N = lane&15
    const int rbase = m0 + ((lane >> 4) << 3);
    #pragma unroll
    for (int j = 0; j < 4; ++j) {
        v8f acc = (j == 0) ? acc0 : (j == 1) ? acc1 : (j == 2) ? acc2 : acc3;
        const int n = n0 + j * 16 + (lane & 15);
        const float bv = bias[n];
        float* cp = Cout + (size_t)rbase * ncols + n;
        #pragma unroll
        for (int r = 0; r < 8; ++r)
            cp[(size_t)r * ncols] = acc[r] + bv;
    }
}

// ---------------------------------------------------------------------------
// Per-column sum / sum-of-squares over N rows (C = 512 fixed).
// Thread t handles cols t and t+256 over rows_per_block rows; atomic combine.
// stats layout: [0..511]=sum, [512..1023]=sumsq
// ---------------------------------------------------------------------------
__global__ __launch_bounds__(256)
void colstats_k(const float* __restrict__ X, float* __restrict__ stats,
                int rows_per_block)
{
    const int c  = threadIdx.x;
    const long r0 = (long)blockIdx.x * rows_per_block;
    float s0 = 0.f, q0 = 0.f, s1 = 0.f, q1 = 0.f;
    for (int r = 0; r < rows_per_block; ++r) {
        const float* row = X + (r0 + r) * 512;
        float v0 = row[c];
        float v1 = row[c + 256];
        s0 += v0; q0 += v0 * v0;
        s1 += v1; q1 += v1 * v1;
    }
    atomicAdd(&stats[c],         s0);
    atomicAdd(&stats[c + 256],   s1);
    atomicAdd(&stats[512 + c],       q0);
    atomicAdd(&stats[512 + c + 256], q1);
}

__global__ void finalize_stats_k(float* __restrict__ stats, float invN)
{
    int c = threadIdx.x;             // one block of 512
    float m = stats[c] * invN;
    float v = stats[512 + c] * invN - m * m;
    stats[c] = m;
    stats[512 + c] = rsqrtf(v + 1e-5f);
}

// ---------------------------------------------------------------------------
// BN (+optional ReLU) and convert to bf16. 4 elems/thread (same row).
// ---------------------------------------------------------------------------
__global__ __launch_bounds__(256)
void bn_act_bf16_k(const float* __restrict__ X, const float* __restrict__ stats,
                   const float* __restrict__ g, const float* __restrict__ be,
                   __bf16* __restrict__ out, int do_relu)
{
    size_t i = ((size_t)blockIdx.x * 256 + threadIdx.x) * 4;
    int c = (int)(i & 511);
    f32x4 v = *(const f32x4*)(X + i);
    bf16x4 o;
    #pragma unroll
    for (int e = 0; e < 4; ++e) {
        int col = c + e;
        float t = (v[e] - stats[col]) * stats[512 + col] * g[col] + be[col];
        if (do_relu) t = fmaxf(t, 0.0f);
        o[e] = (__bf16)t;
    }
    *(bf16x4*)(out + i) = o;
}

// BN3 (no relu) + residual with h1, final ReLU, convert to bf16.
__global__ __launch_bounds__(256)
void bn_residual_bf16_k(const float* __restrict__ Y, const float* __restrict__ stats,
                        const float* __restrict__ g, const float* __restrict__ be,
                        const __bf16* __restrict__ H1, __bf16* __restrict__ out)
{
    size_t i = ((size_t)blockIdx.x * 256 + threadIdx.x) * 4;
    int c = (int)(i & 511);
    f32x4 v = *(const f32x4*)(Y + i);
    bf16x4 h = *(const bf16x4*)(H1 + i);
    bf16x4 o;
    #pragma unroll
    for (int e = 0; e < 4; ++e) {
        int col = c + e;
        float t = (v[e] - stats[col]) * stats[512 + col] * g[col] + be[col];
        t = fmaxf((float)h[e] + t, 0.0f);
        o[e] = (__bf16)t;
    }
    *(bf16x4*)(out + i) = o;
}

// ---------------------------------------------------------------------------
// betas / camera split from padded output (N x 256).
// ---------------------------------------------------------------------------
__global__ __launch_bounds__(256)
void split_out_k(const float* __restrict__ O, float* __restrict__ betas,
                 float* __restrict__ cam, long N)
{
    long i = (long)blockIdx.x * 256 + threadIdx.x;
    if (i >= N * 13) return;
    long n = i / 13;
    int  j = (int)(i - n * 13);
    float v = O[n * 256 + 216 + j];
    if (j < 10) betas[n * 10 + j] = v;
    else        cam[n * 3 + (j - 10)] = v;
}

// ---------------------------------------------------------------------------
// Closest-rotation via scaled Newton polar iteration:
//   X <- 0.5*(z*X + C(X)/(z*det)),  z = |det|^(-1/3);  X^-T = cof(X)/det.
// Converges to U V^T of the SVD; then R *= det(R) as in the reference.
// One 3x3 per thread.
// ---------------------------------------------------------------------------
__global__ __launch_bounds__(256)
void polar_rot_k(const float* __restrict__ O, float* __restrict__ rot, long nmats)
{
    long j = (long)blockIdx.x * 256 + threadIdx.x;
    if (j >= nmats) return;
    long n = j / 24;
    int  q = (int)(j - n * 24);
    const float* src = O + n * 256 + q * 9;

    float x0 = src[0], x1 = src[1], x2 = src[2];
    float x3 = src[3], x4 = src[4], x5 = src[5];
    float x6 = src[6], x7 = src[7], x8 = src[8];

    #pragma unroll
    for (int it = 0; it < 8; ++it) {
        float c00 = x4 * x8 - x5 * x7;
        float c01 = x5 * x6 - x3 * x8;
        float c02 = x3 * x7 - x4 * x6;
        float c10 = x2 * x7 - x1 * x8;
        float c11 = x0 * x8 - x2 * x6;
        float c12 = x1 * x6 - x0 * x7;
        float c20 = x1 * x5 - x2 * x4;
        float c21 = x2 * x3 - x0 * x5;
        float c22 = x0 * x4 - x1 * x3;
        float det = x0 * c00 + x1 * c01 + x2 * c02;
        float ad  = fmaxf(fabsf(det), 1e-30f);
        float z   = 1.0f / cbrtf(ad);        // |det|^(-1/3)
        float wa  = 0.5f * z;
        float wb  = 0.5f / (det * z);
        x0 = wa * x0 + wb * c00;  x1 = wa * x1 + wb * c01;  x2 = wa * x2 + wb * c02;
        x3 = wa * x3 + wb * c10;  x4 = wa * x4 + wb * c11;  x5 = wa * x5 + wb * c12;
        x6 = wa * x6 + wb * c20;  x7 = wa * x7 + wb * c21;  x8 = wa * x8 + wb * c22;
    }

    // det(R) is ~ +/-1; reference multiplies R by it.
    float dr = x0 * (x4 * x8 - x5 * x7)
             - x1 * (x3 * x8 - x5 * x6)
             + x2 * (x3 * x7 - x4 * x6);
    float* dst = rot + j * 9;
    dst[0] = x0 * dr; dst[1] = x1 * dr; dst[2] = x2 * dr;
    dst[3] = x3 * dr; dst[4] = x4 * dr; dst[5] = x5 * dr;
    dst[6] = x6 * dr; dst[7] = x7 * dr; dst[8] = x8 * dr;
}

// ---------------------------------------------------------------------------
extern "C" void kernel_launch(void* const* d_in, const int* in_sizes, int n_in,
                              void* d_out, int out_size, void* d_ws, size_t ws_size,
                              hipStream_t stream)
{
    const float* x   = (const float*)d_in[0];
    const float* W1  = (const float*)d_in[1];
    const float* b1  = (const float*)d_in[2];
    const float* g1  = (const float*)d_in[3];
    const float* be1 = (const float*)d_in[4];
    const float* W2  = (const float*)d_in[5];
    const float* b2  = (const float*)d_in[6];
    const float* g2  = (const float*)d_in[7];
    const float* be2 = (const float*)d_in[8];
    const float* W3  = (const float*)d_in[9];
    const float* b3  = (const float*)d_in[10];
    const float* g3  = (const float*)d_in[11];
    const float* be3 = (const float*)d_in[12];
    const float* W4  = (const float*)d_in[13];
    const float* b4  = (const float*)d_in[14];

    const int  C    = 512;
    const long N    = (long)in_sizes[0] / C;          // 57600
    const int  OUTC = in_sizes[13] / C;               // 229
    const int  OUTP = 256;                            // padded

    // ---- workspace layout (all region sizes multiples of 256B) ----
    char* p = (char*)d_ws;
    float*  RAW  = (float*)p;            p += (size_t)N * C * sizeof(float);     // pre-BN GEMM out
    float*  OUTB = (float*)p;            p += (size_t)N * OUTP * sizeof(float);  // final GEMM out
    __bf16* XB   = (__bf16*)p;           p += (size_t)N * C * sizeof(__bf16);    // x bf16 / reused as H bf16
    __bf16* H1B  = (__bf16*)p;           p += (size_t)N * C * sizeof(__bf16);
    __bf16* H2B  = (__bf16*)p;           p += (size_t)N * C * sizeof(__bf16);
    __bf16* W1B  = (__bf16*)p;           p += (size_t)C * C * sizeof(__bf16);
    __bf16* W2B  = (__bf16*)p;           p += (size_t)C * C * sizeof(__bf16);
    __bf16* W3B  = (__bf16*)p;           p += (size_t)C * C * sizeof(__bf16);
    __bf16* W4B  = (__bf16*)p;           p += (size_t)C * OUTP * sizeof(__bf16);
    float*  B4P  = (float*)p;            p += 256 * sizeof(float);
    float*  STATS = (float*)p;           p += 1024 * sizeof(float);

    const int TPB = 256;
    auto blocks = [](long total) { return (int)((total + 255) / 256); };

    // ---- precision prep: fp32 -> bf16 (W4 padded to 256 cols) ----
    cvt_pad_bf16_k<<<blocks(N * C), TPB, 0, stream>>>(x,  XB,  N, C, C);
    cvt_pad_bf16_k<<<blocks((long)C * C), TPB, 0, stream>>>(W1, W1B, C, C, C);
    cvt_pad_bf16_k<<<blocks((long)C * C), TPB, 0, stream>>>(W2, W2B, C, C, C);
    cvt_pad_bf16_k<<<blocks((long)C * C), TPB, 0, stream>>>(W3, W3B, C, C, C);
    cvt_pad_bf16_k<<<blocks((long)C * OUTP), TPB, 0, stream>>>(W4, W4B, C, OUTC, OUTP);
    pad_bias_k<<<1, 256, 0, stream>>>(b4, B4P, OUTC, OUTP);

    const dim3 gemm_grid((unsigned)(N / 128), C / 64);
    const dim3 gemm_grid4((unsigned)(N / 128), OUTP / 64);
    const int  stat_rows = 256;
    const int  stat_blocks = (int)(N / stat_rows);    // 225
    const int  bn_blocks = (int)(N * C / 4 / TPB);
    const float invN = 1.0f / (float)N;

    // ---- layer 1: h = relu(bn(x @ W1 + b1)) ----
    gemm_bf16_wmma_k<<<gemm_grid, TPB, 0, stream>>>(XB, W1B, b1, RAW, (int)N, C);
    (void)hipMemsetAsync(STATS, 0, 1024 * sizeof(float), stream);
    colstats_k<<<stat_blocks, TPB, 0, stream>>>(RAW, STATS, stat_rows);
    finalize_stats_k<<<1, 512, 0, stream>>>(STATS, invN);
    bn_act_bf16_k<<<bn_blocks, TPB, 0, stream>>>(RAW, STATS, g1, be1, H1B, 1);

    // ---- layer 2: y = relu(bn(h @ W2 + b2)) ----
    gemm_bf16_wmma_k<<<gemm_grid, TPB, 0, stream>>>(H1B, W2B, b2, RAW, (int)N, C);
    (void)hipMemsetAsync(STATS, 0, 1024 * sizeof(float), stream);
    colstats_k<<<stat_blocks, TPB, 0, stream>>>(RAW, STATS, stat_rows);
    finalize_stats_k<<<1, 512, 0, stream>>>(STATS, invN);
    bn_act_bf16_k<<<bn_blocks, TPB, 0, stream>>>(RAW, STATS, g2, be2, H2B, 1);

    // ---- layer 3: y = bn(y @ W3 + b3); h = relu(h + y) ----
    gemm_bf16_wmma_k<<<gemm_grid, TPB, 0, stream>>>(H2B, W3B, b3, RAW, (int)N, C);
    (void)hipMemsetAsync(STATS, 0, 1024 * sizeof(float), stream);
    colstats_k<<<stat_blocks, TPB, 0, stream>>>(RAW, STATS, stat_rows);
    finalize_stats_k<<<1, 512, 0, stream>>>(STATS, invN);
    bn_residual_bf16_k<<<bn_blocks, TPB, 0, stream>>>(RAW, STATS, g3, be3, H1B, XB /*reuse as H*/);

    // ---- layer 4: out = h @ W4 + b4 (padded to 256 cols) ----
    gemm_bf16_wmma_k<<<gemm_grid4, TPB, 0, stream>>>(XB, W4B, B4P, OUTB, (int)N, OUTP);

    // ---- outputs: rotmat | betas | camera concatenated flat ----
    float* rotp  = (float*)d_out;                 // N*24*9
    float* betap = rotp + N * 216;                // N*10
    float* camp  = betap + N * 10;                // N*3
    polar_rot_k<<<blocks(N * 24), TPB, 0, stream>>>(OUTB, rotp, N * 24);
    split_out_k<<<blocks(N * 13), TPB, 0, stream>>>(OUTB, betap, camp, N);
}
